// NLBasicBlock_20675972563414
// MI455X (gfx1250) — matrix operand order repaired and checked by reading, not compiled, not run on hardware
//
#include <hip/hip_runtime.h>
#include <math.h>

typedef __bf16 bf16;
typedef __attribute__((ext_vector_type(16))) __bf16 v16bf;
typedef __attribute__((ext_vector_type(8)))  float  v8f;
typedef __attribute__((ext_vector_type(4)))  int    i4v;

constexpr int kB = 4, kC = 128, kH = 64, kW = 64, kN = kH * kW;   // N = 4096
constexpr size_t kBCN = (size_t)kB * kC * kN;

// --- CDNA5 async global->LDS path (ASYNCcnt), guarded for toolchain support.
#if defined(__has_builtin)
#if __has_builtin(__builtin_amdgcn_global_load_async_to_lds_b128) && \
    __has_builtin(__builtin_amdgcn_s_wait_asynccnt)
#define USE_ASYNC_LDS 1
#endif
#endif
#ifndef USE_ASYNC_LDS
#define USE_ASYNC_LDS 0
#endif

#define AS1 __attribute__((address_space(1)))
#define AS3 __attribute__((address_space(3)))

__device__ __forceinline__ AS3 i4v* lds_i4(void* p) {
  // LDS byte offset == low 32 bits of the generic shared-aperture address
  return (AS3 i4v*)(uint32_t)(uintptr_t)p;
}
__device__ __forceinline__ AS1 i4v* gbl_i4(const void* p) {
  return (AS1 i4v*)(uintptr_t)p;
}

// 16-bit A-matrix 16x32 fragment K index (contiguous, matches B fills):
// lanes 0-15 hold K in {0..15}; lanes 16-31 hold K in {16..31}
__device__ __forceinline__ int afrag_k(int lane, int j) {
  return ((lane < 16) ? 0 : 16) + j;
}

__device__ __forceinline__ v8f wmma_bf16(v16bf a, v16bf b, v8f c) {
  return __builtin_amdgcn_wmma_f32_16x16x32_bf16(false, a, false, b, (short)0, c,
                                                 false, false);
}

// ---------------------------------------------------------------------------
// 3x3 conv (stride 1, pad 1) + PReLU.
// mode 0: out_f32 = conv (pre-activation x1), act_bf16 = bf16(prelu(conv))
// mode 1: out_f32 = prelu(conv)
// Block: 256 threads = 4 out-channels x 8x8 pixel tile. LDS-tiled over ci.
// ---------------------------------------------------------------------------
__global__ __launch_bounds__(256) void conv3x3_prelu_kernel(
    const float* __restrict__ in, const float* __restrict__ wgt,
    const float* __restrict__ prelu_a,
    float* __restrict__ out_f32, bf16* __restrict__ act_bf16, int mode) {
  __shared__ float s_in[8][10][12];   // 8 ci x 10x10 halo tile (padded rows)
  __shared__ float s_w[4][8][9];
  const int tid = threadIdx.x;
  const int b = blockIdx.z;
  const int co0 = blockIdx.y * 4;
  const int tx = (blockIdx.x & 7) * 8, ty = (blockIdx.x >> 3) * 8;
  const int px = tid & 63, lx = px & 7, ly = px >> 3;
  const int col = tid >> 6;
  const float a = prelu_a[0];
  float acc = 0.f;
  for (int cic = 0; cic < 16; ++cic) {
    for (int idx = tid; idx < 800; idx += 256) {
      int ci = idx / 100, r = idx % 100, yy = r / 10, xx = r % 10;
      int gy = ty - 1 + yy, gx = tx - 1 + xx;
      float v = 0.f;
      if (gy >= 0 && gy < kH && gx >= 0 && gx < kW)
        v = in[(((size_t)b * kC + cic * 8 + ci) * kH + gy) * kW + gx];
      s_in[ci][yy][xx] = v;
    }
    for (int idx = tid; idx < 288; idx += 256) {
      int o = idx / 72, r = idx % 72, ci = r / 9, kk = r % 9;
      s_w[o][ci][kk] = wgt[((size_t)(co0 + o) * kC + cic * 8 + ci) * 9 + kk];
    }
    __syncthreads();
#pragma unroll
    for (int ci = 0; ci < 8; ++ci)
#pragma unroll
      for (int kh = 0; kh < 3; ++kh)
#pragma unroll
        for (int kw = 0; kw < 3; ++kw)
          acc = fmaf(s_in[ci][ly + kh][lx + kw], s_w[col][ci][kh * 3 + kw], acc);
    __syncthreads();
  }
  const int n = (ty + ly) * kW + tx + lx;
  const size_t o = ((size_t)b * kC + co0 + col) * kN + n;
  const float pr = acc >= 0.f ? acc : a * acc;
  if (mode == 0) {
    out_f32[o] = acc;
    act_bf16[o] = (bf16)pr;
  } else {
    out_f32[o] = pr;
  }
}

// ---------------------------------------------------------------------------
// theta/phi/g projections: out[o,n] = sum_c W[o,c] * act[c,n]  (bf16 WMMA)
// grid (N/64, B, 3); block = 128 = 4 waves; each wave: 16 n-cols x 128 o-rows.
// ---------------------------------------------------------------------------
__global__ __launch_bounds__(128) void proj_kernel(
    const bf16* __restrict__ act, const float* __restrict__ theta_w,
    const float* __restrict__ phi_w, const float* __restrict__ g_w,
    bf16* __restrict__ theta, bf16* __restrict__ phi, bf16* __restrict__ g) {
  __shared__ bf16 s_w[kC * kC];      // full 128x128 weight, bf16 (32 KB)
  __shared__ bf16 s_bt[4][16 * 32];  // per-wave B tile, [n][c]
  const int tid = threadIdx.x, lane = tid & 31, wave = tid >> 5;
  const int b = blockIdx.y, sel = blockIdx.z;
  const int n0 = blockIdx.x * 64 + wave * 16;
  const float* wsrc = sel == 0 ? theta_w : (sel == 1 ? phi_w : g_w);
  bf16* outp = (sel == 0 ? theta : (sel == 1 ? phi : g)) + (size_t)b * kC * kN;
  const bf16* ab = act + (size_t)b * kC * kN;
  for (int i = tid; i < kC * kC; i += 128) s_w[i] = (bf16)wsrc[i];
  __syncthreads();
  v8f acc[8] = {};
  const int nn = lane & 15, kb = (lane < 16) ? 0 : 16;
  bf16* bt = s_bt[wave];
  for (int ks = 0; ks < 4; ++ks) {
    const int c0 = ks * 32;
    {  // stage bt[n*32+c] = act[c0+c][n0+n]; lane = c row, 32B vector load
      alignas(16) bf16 tmp[16];
      const uint4* src = (const uint4*)(ab + (size_t)(c0 + lane) * kN + n0);
      *(uint4*)(tmp + 0) = src[0];
      *(uint4*)(tmp + 8) = src[1];
#pragma unroll
      for (int t = 0; t < 16; ++t) bt[t * 32 + lane] = tmp[t];
    }
    v16bf bfr;
#pragma unroll
    for (int j = 0; j < 16; ++j) bfr[j] = bt[nn * 32 + kb + j];
#pragma unroll
    for (int ot = 0; ot < 8; ++ot) {
      v16bf afr;
      const bf16* arow = s_w + (ot * 16 + nn) * kC + c0;
#pragma unroll
      for (int j = 0; j < 16; ++j) afr[j] = arow[afrag_k(lane, j)];
      acc[ot] = wmma_bf16(afr, bfr, acc[ot]);
    }
  }
  const int mh = (lane < 16) ? 0 : 8;  // D: M = v + 8*laneHalf, N = lane%16
#pragma unroll
  for (int ot = 0; ot < 8; ++ot)
#pragma unroll
    for (int v = 0; v < 8; ++v)
      outp[(size_t)(ot * 16 + v + mh) * kN + n0 + nn] = (bf16)acc[ot][v];
}

// ---------------------------------------------------------------------------
// Pass 1: column softmax stats of f[n,m] = sum_c theta[c,n]*phi[c,m].
// grid (N/16 m-tiles, B); block 128; online max/sum per column, tiled WMMA.
// ---------------------------------------------------------------------------
__global__ __launch_bounds__(128) void softmax_stats_kernel(
    const bf16* __restrict__ theta, const bf16* __restrict__ phi,
    float* __restrict__ Mbuf, float* __restrict__ Zbuf) {
  __shared__ bf16 s_phi[16 * kC];    // [m][c]
  __shared__ bf16 s_at[4][16 * kC];  // per-wave theta tile [n][c]
  __shared__ float s_mx[128], s_sm[128];
  const int tid = threadIdx.x, lane = tid & 31, wave = tid >> 5;
  const int b = blockIdx.y, m0 = blockIdx.x * 16;
  const bf16* thb = theta + (size_t)b * kC * kN;
  const bf16* phb = phi + (size_t)b * kC * kN;
  {  // stage phi[m0..m0+16) columns: s_phi[m*128+c] = phi[c][m0+m]
    alignas(16) bf16 tmp[16];
    const uint4* src = (const uint4*)(phb + (size_t)tid * kN + m0);
    *(uint4*)(tmp + 0) = src[0];
    *(uint4*)(tmp + 8) = src[1];
#pragma unroll
    for (int t = 0; t < 16; ++t) s_phi[t * kC + tid] = tmp[t];
  }
  __syncthreads();
  const int nn = lane & 15, kb = (lane < 16) ? 0 : 16;
  v16bf bphi[4];  // resident B fragments (K = c)
#pragma unroll
  for (int k = 0; k < 4; ++k)
#pragma unroll
    for (int j = 0; j < 16; ++j) bphi[k][j] = s_phi[nn * kC + k * 32 + kb + j];
  float mx = -INFINITY, sm = 0.f;
  bf16* at = s_at[wave];
  for (int nt = wave; nt < kN / 16; nt += 4) {
    const int n0 = nt * 16;
#pragma unroll
    for (int cb = 0; cb < 4; ++cb) {  // stage theta tile [16n][128c]
      const int cc = cb * 32 + lane;
      alignas(16) bf16 tmp[16];
      const uint4* src = (const uint4*)(thb + (size_t)cc * kN + n0);
      *(uint4*)(tmp + 0) = src[0];
      *(uint4*)(tmp + 8) = src[1];
#pragma unroll
      for (int t = 0; t < 16; ++t) at[t * kC + cc] = tmp[t];
    }
    v8f f = {};
#pragma unroll
    for (int k = 0; k < 4; ++k) {
      v16bf afr;
      const bf16* arow = at + nn * kC + k * 32;
#pragma unroll
      for (int j = 0; j < 16; ++j) afr[j] = arow[afrag_k(lane, j)];
      f = wmma_bf16(afr, bphi[k], f);
    }
    float tmx = f[0];
#pragma unroll
    for (int v = 1; v < 8; ++v) tmx = fmaxf(tmx, f[v]);
    const float nmx = fmaxf(mx, tmx);
    float ls = 0.f;
#pragma unroll
    for (int v = 0; v < 8; ++v) ls += __expf(f[v] - nmx);
    sm = sm * __expf(mx - nmx) + ls;
    mx = nmx;
  }
  s_mx[tid] = mx;
  s_sm[tid] = sm;
  __syncthreads();
  if (tid < 16) {  // column m0+tid lives in lanes {tid, tid+16} of each wave
    float M = -INFINITY;
#pragma unroll
    for (int w = 0; w < 4; ++w) {
      M = fmaxf(M, s_mx[w * 32 + tid]);
      M = fmaxf(M, s_mx[w * 32 + tid + 16]);
    }
    float Z = 0.f;
#pragma unroll
    for (int w = 0; w < 4; ++w) {
      Z += s_sm[w * 32 + tid] * __expf(s_mx[w * 32 + tid] - M);
      Z += s_sm[w * 32 + tid + 16] * __expf(s_mx[w * 32 + tid + 16] - M);
    }
    Mbuf[(size_t)b * kN + m0 + tid] = M;
    Zbuf[(size_t)b * kN + m0 + tid] = Z;
  }
}

// ---------------------------------------------------------------------------
// Pass 2: y[c,n] = sum_m exp(f[n,m]-M[m])/Z[m] * g[c,m].
// grid (N/64, B); block 128; wave owns a 16-row n-tile and all 128 c.
// Recomputes f tiles (flash-style). g tile staged with ASYNC global->LDS
// copies (both sides contiguous 64B/row), overlapped with the phi transpose staging.
// ---------------------------------------------------------------------------
__global__ __launch_bounds__(128) void attn_y_kernel(
    const bf16* __restrict__ theta, const bf16* __restrict__ phi,
    const bf16* __restrict__ g, const float* __restrict__ Mbuf,
    const float* __restrict__ Zbuf, float* __restrict__ y) {
  __shared__ bf16 s_phi[32 * kC];    // [m][c]
  __shared__ bf16 s_g[kC * 32];      // [c][m]
  __shared__ bf16 s_at[4][16 * kC];  // per-wave theta tile
  __shared__ bf16 s_p[4][16 * 32];   // per-wave P tile [n][m]
  const int tid = threadIdx.x, lane = tid & 31, wave = tid >> 5;
  const int b = blockIdx.y;
  const int n0 = blockIdx.x * 64 + wave * 16;
  const bf16* thb = theta + (size_t)b * kC * kN;
  const bf16* phb = phi + (size_t)b * kC * kN;
  const bf16* gbp = g + (size_t)b * kC * kN;
  const float* Mb = Mbuf + (size_t)b * kN;
  const float* Zb = Zbuf + (size_t)b * kN;
  const int nn = lane & 15, kb = (lane < 16) ? 0 : 16, mh = (lane < 16) ? 0 : 8;
  bf16* at = s_at[wave];
#pragma unroll
  for (int cb = 0; cb < 4; ++cb) {  // stage theta tile once (n fixed per wave)
    const int cc = cb * 32 + lane;
    alignas(16) bf16 tmp[16];
    const uint4* src = (const uint4*)(thb + (size_t)cc * kN + n0);
    *(uint4*)(tmp + 0) = src[0];
    *(uint4*)(tmp + 8) = src[1];
#pragma unroll
    for (int t = 0; t < 16; ++t) at[t * kC + cc] = tmp[t];
  }
  v16bf ath[4];  // resident theta A fragments
#pragma unroll
  for (int k = 0; k < 4; ++k) {
    const bf16* arow = at + nn * kC + k * 32;
#pragma unroll
    for (int j = 0; j < 16; ++j) ath[k][j] = arow[afrag_k(lane, j)];
  }
  v8f yacc[8] = {};
  bf16* pt = s_p[wave];
  for (int ms = 0; ms < kN / 32; ++ms) {
    const int m0 = ms * 32;
    __syncthreads();
    {  // stage g [128c][32m]: contiguous both sides -> async global->LDS
      const bf16* gsrc = gbp + (size_t)tid * kN + m0;
      bf16* gdst = s_g + tid * 32;
#if USE_ASYNC_LDS
#pragma unroll
      for (int q = 0; q < 4; ++q)
        __builtin_amdgcn_global_load_async_to_lds_b128(
            gbl_i4(gsrc + q * 8), lds_i4(gdst + q * 8), 0, 0);
#else
      const uint4* sg = (const uint4*)gsrc;
      uint4* dg = (uint4*)gdst;
#pragma unroll
      for (int q = 0; q < 4; ++q) dg[q] = sg[q];
#endif
      // stage phi [32m][128c] (transpose -> manual path)
      alignas(16) bf16 tmp[32];
      const uint4* sp = (const uint4*)(phb + (size_t)tid * kN + m0);
#pragma unroll
      for (int q = 0; q < 4; ++q) *(uint4*)(tmp + q * 8) = sp[q];
#pragma unroll
      for (int t = 0; t < 32; ++t) s_phi[t * kC + tid] = tmp[t];
    }
#if USE_ASYNC_LDS
    __builtin_amdgcn_s_wait_asynccnt(0);  // this wave's async copies landed
#endif
    __syncthreads();
#pragma unroll
    for (int msub = 0; msub < 2; ++msub) {
      v8f f = {};
#pragma unroll
      for (int k = 0; k < 4; ++k) {
        v16bf bfr;
#pragma unroll
        for (int j = 0; j < 16; ++j)
          bfr[j] = s_phi[(msub * 16 + nn) * kC + k * 32 + kb + j];
        f = wmma_bf16(ath[k], bfr, f);
      }
      const int mcol = m0 + msub * 16 + nn;  // D lane column == softmax column
      const float Mv = Mb[mcol];
      const float rZ = 1.0f / Zb[mcol];
#pragma unroll
      for (int v = 0; v < 8; ++v)
        pt[(v + mh) * 32 + msub * 16 + nn] = (bf16)(__expf(f[v] - Mv) * rZ);
    }
    v16bf pa;  // P as A fragment: M = n, K = m
#pragma unroll
    for (int j = 0; j < 16; ++j) pa[j] = pt[nn * 32 + afrag_k(lane, j)];
#pragma unroll
    for (int ct = 0; ct < 8; ++ct) {
      v16bf gfr;  // B: K = m, N = c
#pragma unroll
      for (int j = 0; j < 16; ++j) gfr[j] = s_g[(ct * 16 + nn) * 32 + kb + j];
      yacc[ct] = wmma_bf16(pa, gfr, yacc[ct]);
    }
  }
#pragma unroll
  for (int ct = 0; ct < 8; ++ct) {  // D: rows = n (contiguous over v)
    float* yp = y + ((size_t)b * kC + ct * 16 + nn) * kN + n0 + mh;
    ((float4*)yp)[0] = make_float4(yacc[ct][0], yacc[ct][1], yacc[ct][2], yacc[ct][3]);
    ((float4*)yp)[1] = make_float4(yacc[ct][4], yacc[ct][5], yacc[ct][6], yacc[ct][7]);
  }
}

// ---------------------------------------------------------------------------
// res = W_w @ y + prelu(x1) + x1   (nonlocal output + both residuals)
// ---------------------------------------------------------------------------
__global__ __launch_bounds__(128) void wproj_res_kernel(
    const float* __restrict__ yf, const float* __restrict__ Ww,
    const float* __restrict__ x1, const float* __restrict__ prelu_a,
    float* __restrict__ res) {
  __shared__ bf16 s_w[kC * kC];
  __shared__ bf16 s_bt[4][16 * 32];
  const int tid = threadIdx.x, lane = tid & 31, wave = tid >> 5;
  const int b = blockIdx.y;
  const int n0 = blockIdx.x * 64 + wave * 16;
  const float a = prelu_a[0];
  for (int i = tid; i < kC * kC; i += 128) s_w[i] = (bf16)Ww[i];
  __syncthreads();
  const float* yb = yf + (size_t)b * kC * kN;
  v8f acc[8] = {};
  const int nn = lane & 15, kb = (lane < 16) ? 0 : 16;
  bf16* bt = s_bt[wave];
  for (int ks = 0; ks < 4; ++ks) {
    const int c0 = ks * 32;
    {
      alignas(16) float tmp[16];
      const float4* src = (const float4*)(yb + (size_t)(c0 + lane) * kN + n0);
#pragma unroll
      for (int q = 0; q < 4; ++q) *(float4*)(tmp + q * 4) = src[q];
#pragma unroll
      for (int t = 0; t < 16; ++t) bt[t * 32 + lane] = (bf16)tmp[t];
    }
    v16bf bfr;
#pragma unroll
    for (int j = 0; j < 16; ++j) bfr[j] = bt[nn * 32 + kb + j];
#pragma unroll
    for (int ot = 0; ot < 8; ++ot) {
      v16bf afr;
      const bf16* arow = s_w + (ot * 16 + nn) * kC + c0;
#pragma unroll
      for (int j = 0; j < 16; ++j) afr[j] = arow[afrag_k(lane, j)];
      acc[ot] = wmma_bf16(afr, bfr, acc[ot]);
    }
  }
  const int mh = (lane < 16) ? 0 : 8;
#pragma unroll
  for (int ot = 0; ot < 8; ++ot)
#pragma unroll
    for (int v = 0; v < 8; ++v) {
      const size_t idx = ((size_t)b * kC + ot * 16 + v + mh) * kN + n0 + nn;
      const float xv = x1[idx];
      const float pr = xv >= 0.f ? xv : a * xv;
      res[idx] = acc[ot][v] + xv + pr;
    }
}

extern "C" void kernel_launch(void* const* d_in, const int* in_sizes, int n_in,
                              void* d_out, int out_size, void* d_ws, size_t ws_size,
                              hipStream_t stream) {
  const float* x       = (const float*)d_in[0];
  const float* conv1_w = (const float*)d_in[1];
  const float* prelu_a = (const float*)d_in[2];
  const float* theta_w = (const float*)d_in[3];
  const float* phi_w   = (const float*)d_in[4];
  const float* g_w     = (const float*)d_in[5];
  const float* W_w     = (const float*)d_in[6];
  const float* conv2_w = (const float*)d_in[7];

  char* p = (char*)d_ws;                      // ~42 MB total scratch
  float* x1 = (float*)p;  p += kBCN * 4;
  bf16* act = (bf16*)p;   p += kBCN * 2;
  bf16* th  = (bf16*)p;   p += kBCN * 2;
  bf16* ph  = (bf16*)p;   p += kBCN * 2;
  bf16* gg  = (bf16*)p;   p += kBCN * 2;
  float* Mb = (float*)p;  p += (size_t)kB * kN * 4;
  float* Zb = (float*)p;  p += (size_t)kB * kN * 4;
  float* yb = (float*)p;  p += kBCN * 4;
  float* res = (float*)p; p += kBCN * 4;

  conv3x3_prelu_kernel<<<dim3(64, 32, 4), 256, 0, stream>>>(
      x, conv1_w, prelu_a, x1, act, 0);
  proj_kernel<<<dim3(64, 4, 3), 128, 0, stream>>>(
      act, theta_w, phi_w, g_w, th, ph, gg);
  softmax_stats_kernel<<<dim3(256, 4), 128, 0, stream>>>(th, ph, Mb, Zb);
  attn_y_kernel<<<dim3(64, 4), 128, 0, stream>>>(th, ph, gg, Mb, Zb, yb);
  wproj_res_kernel<<<dim3(64, 4), 128, 0, stream>>>(yb, W_w, x1, prelu_a, res);
  conv3x3_prelu_kernel<<<dim3(64, 32, 4), 256, 0, stream>>>(
      res, conv2_w, prelu_a, (float*)d_out, nullptr, 1);
}
